// DownscaleLabel_18614388261624
// MI455X (gfx1250) — compile-verified, matrix operand order checked by reading, not run on hardware
//
#include <hip/hip_runtime.h>
#include <hip/hip_bf16.h>

typedef int v8i __attribute__((ext_vector_type(8)));
typedef int v4i __attribute__((ext_vector_type(4)));

#define AS1 __attribute__((address_space(1)))
#define AS3 __attribute__((address_space(3)))

#if defined(__has_builtin)
#  if __has_builtin(__builtin_amdgcn_global_load_async_to_lds_b128)
#    define USE_ASYNC_LDS 1
#  endif
#  if __has_builtin(__builtin_amdgcn_s_wait_asynccnt)
#    define HAVE_WAIT_ASYNC 1
#  endif
#endif

// 128 dwords per strip row + 4 pad dwords: row stride 132 ≡ 4 (mod 64 banks),
// so the 16 per-row b128 reads cover all 64 banks conflict-free, and stays
// 0 mod 4 dwords so b128 LDS accesses remain 16B-aligned.
#define LDS_STRIDE 132

// One workgroup (256 threads = 8 wave32) handles a 16-row x 128-col strip
// = 8 adjacent 16x16 blocks; one wave per block.
// Histogram per block via V_WMMA_I32_16X16X64_IU8:
//   A[m][k] (16x64 u8): row m = block row; K packs 8 elements x 8-byte one-hot.
//   B[k][n] (64x16 u8): constant, B[k][n] = (k % 8 == n && n < 8).
// The column->K-slot assignment is free (WMMA sums over K; B depends only on
// k mod 8), so lanes 0-15 take 4 consecutive cols and lanes 16-31 the next 4:
// one ds_load_b128 per lane per WMMA. Two independent WMMAs (separate
// accumulators -> no IU8 D->C RAW chain) cover all 16 columns.
__global__ __launch_bounds__(256) void downscale_label_kernel(
    const int* __restrict__ label, int* __restrict__ out)
{
    const int strip = blockIdx.x;
    const int sx = strip & 7;          // strip column (8 strips of 128 cols)
    const int by = (strip >> 3) & 63;  // block row
    const int bi = strip >> 9;         // batch

    __shared__ int lds[16 * LDS_STRIDE];

    const int t    = threadIdx.x;
    const int lane = t & 31;
    const int wave = t >> 5;

    const size_t rowbase = ((size_t)bi * 64 + by) * 16;  // first image row of strip
    const int    colbase = sx * 128;                     // first dword col of strip

    // ---- Stage 16x128 int32 strip into LDS (async copy, coalesced b128) ----
    {
        const int r = t >> 5;          // 0..7
        const int c = (t & 31) * 4;    // dword col within strip (16B chunks)
        const int* g0 = label + (rowbase + r)     * 1024 + colbase + c;
        const int* g1 = label + (rowbase + r + 8) * 1024 + colbase + c;
        int* l0 = &lds[r       * LDS_STRIDE + c];
        int* l1 = &lds[(r + 8) * LDS_STRIDE + c];
#if defined(USE_ASYNC_LDS)
        __builtin_amdgcn_global_load_async_to_lds_b128((AS1 v4i*)g0, (AS3 v4i*)l0, 0, 0);
        __builtin_amdgcn_global_load_async_to_lds_b128((AS1 v4i*)g1, (AS3 v4i*)l1, 0, 0);
#  if defined(HAVE_WAIT_ASYNC)
        __builtin_amdgcn_s_wait_asynccnt(0);
#  else
        asm volatile("s_wait_asynccnt 0" ::: "memory");
#  endif
#else
        *(v4i*)l0 = *(const v4i*)g0;
        *(v4i*)l1 = *(const v4i*)g1;
#endif
    }
    __syncthreads();

    // ---- Per-wave block histogram via WMMA IU8 ----
    const int r    = lane & 15;   // block row handled by this lane (A: M = lane%16)
    const int half = lane >> 4;   // which 4-column group of each 8-col half
    const int bcol = wave * 16;   // this wave's block within the strip

    // Constant B: VGPR v holds K = 4v..4v+3 (lane<16) / 32+4v..35+4v (lane>=16);
    // since 32 % 8 == 0 the byte pattern is identical for both halves.
    v8i Bm;
    {
        const int n = lane & 15;
        #pragma unroll
        for (int v = 0; v < 8; ++v) {
            const int bpos = (n - 4 * v) & 7;
            Bm[v] = (n < 8 && bpos < 4) ? (1 << (8 * bpos)) : 0;
        }
    }

    // 4 consecutive labels per lane per WMMA, via one b128 LDS read each.
    // Labels in [-1,7): lab*8 masked mod 64 by the hardware 64-bit shift maps
    // -1 -> byte 7 (ignore class) and 0..6 to their own byte.
    const v4i q0 = *(const v4i*)&lds[r * LDS_STRIDE + bcol + half * 4];
    const v4i q1 = *(const v4i*)&lds[r * LDS_STRIDE + bcol + 8 + half * 4];

    v8i A0, A1;
    #pragma unroll
    for (int e = 0; e < 4; ++e) {
        const unsigned long long oh0 = 1ull << (q0[e] * 8);
        const unsigned long long oh1 = 1ull << (q1[e] * 8);
        A0[2 * e]     = (int)(unsigned)(oh0 & 0xffffffffu);
        A0[2 * e + 1] = (int)(unsigned)(oh0 >> 32);
        A1[2 * e]     = (int)(unsigned)(oh1 & 0xffffffffu);
        A1[2 * e + 1] = (int)(unsigned)(oh1 >> 32);
    }

    const v8i zero = {};
    // Two independent WMMAs: no D->C RAW between them (IU8 hazard = 9 NOPs
    // avoided); the VALU reduction of acc0 fills acc1's coexec window.
    v8i acc0 = __builtin_amdgcn_wmma_i32_16x16x64_iu8(
        false, A0, false, Bm, zero, false, false);
    v8i acc1 = __builtin_amdgcn_wmma_i32_16x16x64_iu8(
        false, A1, false, Bm, zero, false, false);

    // ---- Reduce C over M (8 VGPRs x 2 accs) + fold lane halves ----
    int s0 = acc0[0] + acc0[1] + acc0[2] + acc0[3]
           + acc0[4] + acc0[5] + acc0[6] + acc0[7];
    int s1 = acc1[0] + acc1[1] + acc1[2] + acc1[3]
           + acc1[4] + acc1[5] + acc1[6] + acc1[7];
    int s = s0 + s1;
    s += __shfl_xor(s, 16, 32);
    // Now lane n (n = lane & 15) holds count of class n for this block (n < 8;
    // classes 8..15 count 0 by construction of B). Halves are identical, so
    // the reduction below only needs offsets 8,4,2,1.

    // ---- First-argmax via packed max-key reduction ----
    // key = count*32 + (31 - class): max key picks the largest count and, on
    // ties, the smallest class index (matches jnp.argmax first-max).
    int key = (s << 5) | (31 - (lane & 15));
    #pragma unroll
    for (int off = 8; off >= 1; off >>= 1) {
        const int other = __shfl_xor(key, off, 32);
        key = (other > key) ? other : key;
    }
    const int best_v = key >> 5;
    const int best_c = 31 - (key & 31);

    int res = (best_c == 7) ? -1 : best_c;   // dominant class was 'ignore'
    if (best_v < 192) res = -1;              // max ratio < 0.75 (192 = 0.75*256)

    if (lane == 0) {
        out[((size_t)bi * 64 + by) * 64 + (sx * 8 + wave)] = res;
    }
}

extern "C" void kernel_launch(void* const* d_in, const int* in_sizes, int n_in,
                              void* d_out, int out_size, void* d_ws, size_t ws_size,
                              hipStream_t stream) {
    (void)in_sizes; (void)n_in; (void)out_size; (void)d_ws; (void)ws_size;
    const int* label = (const int*)d_in[0];
    int* out = (int*)d_out;
    // 16 batches * 64 block-rows * 8 strips = 8192 workgroups of 256 threads
    dim3 grid(16 * 64 * 8);
    dim3 block(256);
    hipLaunchKernelGGL(downscale_label_kernel, grid, block, 0, stream, label, out);
}
